// ECC_35742717838042
// MI455X (gfx1250) — matrix operand order repaired and checked by reading, not compiled, not run on hardware
//
#include <hip/hip_runtime.h>

#define DD   32          // embedding dim
#define HH   16          // hidden width of edge MLP
#define KT   17          // K-tiles: 16 hidden + 1 bias row group
#define EPSV 1e-5f

typedef __attribute__((ext_vector_type(16))) _Float16 v16h;
typedef __attribute__((ext_vector_type(8)))  float    v8f;

// ---------------------------------------------------------------------------
// Pack F = [f_w2 (16 rows); f_b2 (bias row)] -> 17 K-tiles x 2 N-tiles of
// WMMA B fragments (f16, 32x16 K-major per tile) in the gfx1250 wave32 layout:
//   lane 0-15  : N = lane,    K = {0..7, 16..23}   (j = half index 0..15)
//   lane 16-31 : N = lane-16, K = {8..15, 24..31}
// Element (tile t, ntile nt) value = Fmat[t*32 + K][nt*16 + N], where
// Fmat[k][i][o] = (k<16) ? f_w2[k][i*32+o] : f_b2[i*32+o].
// ---------------------------------------------------------------------------
__global__ void pack_bfrag_k(const float* __restrict__ fw2,
                             const float* __restrict__ fb2,
                             _Float16* __restrict__ bfrag) {
  int idx = blockIdx.x * blockDim.x + threadIdx.x;
  if (idx >= KT * 2 * 32 * 16) return;
  int j    = idx & 15;
  int lane = (idx >> 4) & 31;
  int nt   = (idx >> 9) & 1;
  int t    = idx >> 10;
  int K = ((lane >> 4) << 3) + (j & 7) + ((j >> 3) << 4);  // input-feature index i
  int o = nt * 16 + (lane & 15);             // output-feature index
  float v = (t < 16) ? fw2[t * (DD * DD) + K * DD + o]
                     : fb2[K * DD + o];
  bfrag[idx] = (_Float16)v;
}

// ---------------------------------------------------------------------------
// y[n][d] = bias[d] + sum_i x[n][i] * root[i][d]; also zero BN stats buffer.
// ---------------------------------------------------------------------------
__global__ void node_init_k(const float* __restrict__ x,
                            const float* __restrict__ root,
                            const float* __restrict__ bias,
                            float* __restrict__ y,
                            float* __restrict__ stats,
                            int N) {
  if (blockIdx.x == 0 && threadIdx.x < 64) stats[threadIdx.x] = 0.0f;
  int idx = blockIdx.x * blockDim.x + threadIdx.x;
  if (idx >= N * DD) return;
  int n = idx >> 5, d = idx & 31;
  const float* xr = x + n * DD;
  float acc = bias[d];
  #pragma unroll
  for (int i = 0; i < DD; ++i) acc += xr[i] * root[i * DD + d];
  y[idx] = acc;
}

// ---------------------------------------------------------------------------
// Fused edge-MLP + per-edge matvec + scatter-add, as one GEMM:
//   msg[e][o] = sum_{k,i} c[e][k] * x[src[e]][i] * Fmat[k][i][o]
// One wave handles a tile of 16 edges. A-fragment for K-step t is the
// rank-1 product c_t * xvec built in registers; 2 WMMAs (N=0..15, 16..31)
// accumulate in f32. D tile scattered with global f32 atomics.
// ---------------------------------------------------------------------------
__global__ void __launch_bounds__(256)
ecc_edge_wmma_k(const float* __restrict__ x,
                const int*   __restrict__ src,
                const int*   __restrict__ dst,
                const float* __restrict__ ea,
                const float* __restrict__ w1,
                const float* __restrict__ b1,
                const _Float16* __restrict__ bfrag,
                float* __restrict__ y,
                int E) {
  const int lane   = threadIdx.x & 31;
  const int row    = lane & 15;        // edge row (M) within tile
  const int hsel   = lane >> 4;        // lane half: selects K/M sub-ranges
  const int wavesPerBlk = blockDim.x >> 5;
  int wave   = blockIdx.x * wavesPerBlk + (threadIdx.x >> 5);
  int nwaves = gridDim.x * wavesPerBlk;
  int ntiles = (E + 15) >> 4;

  for (int blk = wave; blk < ntiles; blk += nwaves) {
    int base = blk << 4;
    int e    = base + row;
    float valid = (e < E) ? 1.0f : 0.0f;
    int ec = (e < E) ? e : (E - 1);

    // Gather source-node row in A-matrix lane layout (ISA 16-bit A 16x32):
    //   lanes 0-15 : i = {0..7, 16..23};  lanes 16-31 : i = {8..15, 24..31}
    int s = src[ec];
    const float4* xr = (const float4*)(x + s * DD);
    float4 f0 = xr[hsel * 2 + 0];
    float4 f1 = xr[hsel * 2 + 1];
    float4 f2 = xr[hsel * 2 + 4];
    float4 f3 = xr[hsel * 2 + 5];
    v16h xv;
    xv[0]=(_Float16)f0.x;  xv[1]=(_Float16)f0.y;  xv[2]=(_Float16)f0.z;  xv[3]=(_Float16)f0.w;
    xv[4]=(_Float16)f1.x;  xv[5]=(_Float16)f1.y;  xv[6]=(_Float16)f1.z;  xv[7]=(_Float16)f1.w;
    xv[8]=(_Float16)f2.x;  xv[9]=(_Float16)f2.y;  xv[10]=(_Float16)f2.z; xv[11]=(_Float16)f2.w;
    xv[12]=(_Float16)f3.x; xv[13]=(_Float16)f3.y; xv[14]=(_Float16)f3.z; xv[15]=(_Float16)f3.w;

    float a = ea[ec];

    v8f acc0 = {};
    v8f acc1 = {};

    #pragma unroll
    for (int t = 0; t < KT; ++t) {
      // edge-MLP hidden unit t (uniform w1/b1 loads -> scalar path)
      float c = (t < 16) ? fmaxf(a * w1[t] + b1[t], 0.0f) : 1.0f;
      c *= valid;
      _Float16 ch = (_Float16)c;
      v16h cv;
      #pragma unroll
      for (int j = 0; j < 16; ++j) cv[j] = ch;
      v16h af = cv * xv;                       // A fragment = c_t * x_src

      const v16h* bp = ((const v16h*)bfrag) + (t * 2) * 32 + lane;
      v16h bf0 = bp[0];
      v16h bf1 = bp[32];

      acc0 = __builtin_amdgcn_wmma_f32_16x16x32_f16(
          false, af, false, bf0, (short)0, acc0, false, false);
      acc1 = __builtin_amdgcn_wmma_f32_16x16x32_f16(
          false, af, false, bf1, (short)0, acc1, false, false);
    }

    // Scatter-add the 16x32 message tile. C/D layout: lane<16 -> N=lane,
    // M=r; lane>=16 -> N=lane-16, M=r+8.
    int mbase = hsel * 8;
    int ocol  = row;
    #pragma unroll
    for (int r = 0; r < 8; ++r) {
      int ee = base + mbase + r;
      if (ee >= E) ee = E - 1;                 // masked rows add exact 0.0f
      int de = dst[ee];
      atomicAdd(&y[de * DD + ocol],      acc0[r]);
      atomicAdd(&y[de * DD + 16 + ocol], acc1[r]);
    }
  }
}

// ---------------------------------------------------------------------------
// Per-column sum / sum-of-squares of relu(y) over nodes (training BN stats).
// ---------------------------------------------------------------------------
__global__ void bn_stats_k(const float* __restrict__ y,
                           float* __restrict__ stats, int N) {
  __shared__ float ssum[256];
  __shared__ float ssq[256];
  int tid = threadIdx.x;
  int d = tid & 31;
  int g = tid >> 5;                            // 8 row-groups per block
  float s = 0.0f, q = 0.0f;
  for (int n = blockIdx.x * 8 + g; n < N; n += gridDim.x * 8) {
    float v = fmaxf(y[n * DD + d], 0.0f);
    s += v; q += v * v;
  }
  ssum[tid] = s; ssq[tid] = q;
  __syncthreads();
  if (tid < 32) {
    #pragma unroll
    for (int r = 1; r < 8; ++r) { s += ssum[tid + 32 * r]; q += ssq[tid + 32 * r]; }
    atomicAdd(&stats[d], s);
    atomicAdd(&stats[32 + d], q);
  }
}

// ---------------------------------------------------------------------------
// out = (relu(y) - mean) * rsqrt(var + eps) * gamma + beta
// ---------------------------------------------------------------------------
__global__ void bn_apply_k(const float* __restrict__ y,
                           const float* __restrict__ stats,
                           const float* __restrict__ gamma,
                           const float* __restrict__ beta,
                           float* __restrict__ out, int N) {
  int idx = blockIdx.x * blockDim.x + threadIdx.x;
  if (idx >= N * DD) return;
  int d = idx & 31;
  float invN = 1.0f / (float)N;
  float m   = stats[d] * invN;
  float var = stats[32 + d] * invN - m * m;
  float r = fmaxf(y[idx], 0.0f);
  out[idx] = (r - m) * rsqrtf(var + EPSV) * gamma[d] + beta[d];
}

// ---------------------------------------------------------------------------
extern "C" void kernel_launch(void* const* d_in, const int* in_sizes, int n_in,
                              void* d_out, int out_size, void* d_ws, size_t ws_size,
                              hipStream_t stream) {
  (void)n_in; (void)out_size; (void)ws_size;
  const float* x_in = (const float*)d_in[0];
  const int*   eix  = (const int*)d_in[1];
  const float* ea   = (const float*)d_in[2];
  int N = in_sizes[0] / DD;
  int E = in_sizes[1] / 2;
  const int* src = eix;
  const int* dst = eix + E;

  // workspace: y | xbuf | stats(64) | bfrag (64B-aligned)
  float* y     = (float*)d_ws;
  float* xbuf  = y + (size_t)N * DD;
  float* stats = xbuf + (size_t)N * DD;
  size_t boff = ((size_t)((char*)(stats + 64) - (char*)d_ws) + 63) & ~(size_t)63;
  _Float16* bfrag = (_Float16*)((char*)d_ws + boff);

  float* out = (float*)d_out;
  const float* xcur = x_in;

  const int threads = 256;
  int nElem = N * DD;

  for (int l = 0; l < 3; ++l) {
    const float* fw1  = (const float*)d_in[3 + l * 8 + 0];
    const float* fb1  = (const float*)d_in[3 + l * 8 + 1];
    const float* fw2  = (const float*)d_in[3 + l * 8 + 2];
    const float* fb2  = (const float*)d_in[3 + l * 8 + 3];
    const float* root = (const float*)d_in[3 + l * 8 + 4];
    const float* bias = (const float*)d_in[3 + l * 8 + 5];
    const float* gam  = (const float*)d_in[3 + l * 8 + 6];
    const float* bet  = (const float*)d_in[3 + l * 8 + 7];
    float* xnext = (l == 2) ? out : xbuf;

    pack_bfrag_k<<<(KT * 2 * 32 * 16 + threads - 1) / threads, threads, 0, stream>>>(
        fw2, fb2, bfrag);
    node_init_k<<<(nElem + threads - 1) / threads, threads, 0, stream>>>(
        xcur, root, bias, y, stats, N);
    ecc_edge_wmma_k<<<320, threads, 0, stream>>>(
        xcur, src, dst, ea, fw1, fb1, bfrag, y, E);
    bn_stats_k<<<128, threads, 0, stream>>>(y, stats, N);
    bn_apply_k<<<(nElem + threads - 1) / threads, threads, 0, stream>>>(
        y, stats, gam, bet, xnext, N);

    xcur = xnext;
  }
}
